// TraceModule_35278861369502
// MI455X (gfx1250) — matrix operand order repaired and checked by reading, not compiled, not run on hardware
//
#include <hip/hip_runtime.h>
#include <hip/hip_bf16.h>
#include <stdint.h>

typedef __attribute__((ext_vector_type(16))) _Float16 v16h;
typedef __attribute__((ext_vector_type(8)))  float    v8f;

#define HDIM 64
#define WAVES_PER_BLOCK 4
#define RAYS_PER_BLOCK  (WAVES_PER_BLOCK * 16)
#define SDF_THR 5e-5f
#define BIGT 1e10f

__device__ __forceinline__ void wait_ds() {
    asm volatile("s_wait_dscnt 0" ::: "memory");
}

// Fast transcendentals: single v_exp_f32 / v_log_f32 / v_rcp_f32 TRANS ops
// (TRANS co-executes with WMMA per ISA scheduling notes); with f16 activation
// storage their ~1ulp error is irrelevant.
__device__ __forceinline__ float softplusf(float x) {
    return fmaxf(x, 0.f) + __logf(1.f + __expf(-fabsf(x)));
}
__device__ __forceinline__ float sigmoidf_(float x) {
    return __builtin_amdgcn_rcpf(1.f + __expf(-x));
}
__device__ __forceinline__ float ftanh(float x) {
    float e = __expf(-2.f * fabsf(x));
    float t = (1.f - e) * __builtin_amdgcn_rcpf(1.f + e);
    return copysignf(t, x);
}

union FragU {            // one WMMA 16-half fragment per lane
    uint32_t u[8];
    v16h     h;
};
union H2 { uint32_t u; _Float16 h[2]; };

struct WaveCtx {
    const float* W1;     // [3][64]
    const float* b1;     // [64]
    const float* b2;     // [64]
    const float* W3;     // [64]
    float        b3;
    const FragU* W2B;    // [(tile*2+ks)*32 + lane]  B-frags of W2
    const FragU* W2TB;   // same layout, W2^T
    const FragU* W3B;    // [ks*32+lane] col0 = W3, rest 0
    const FragU* W1TB;   // [ks*32+lane] cols0..2 = W1^T, rest 0
    float*       pts;    // [16][4] (reused as gradient scatter buffer)
    float*       red;    // [16] layer-3 reduction bounce
    _Float16*    h1;     // 16*64
    _Float16*    h2;     // 16*64
    _Float16*    sp1;    // 16*64 sigmoid(a1)
    _Float16*    sp2;    // 16*64 sigmoid(a2)
    int          lane;
};

// A-matrix fragment in contiguous k order (k = 16*hi + s). Per lane this is one
// contiguous 32B chunk of a row -> compiler emits ds_load_b128 x2.
__device__ __forceinline__ v16h load_a_frag(const _Float16* hbuf, int ks, int lane) {
    FragU a;
    int m  = lane & 15;
    int hi = lane >> 4;
#pragma unroll
    for (int v = 0; v < 8; ++v) {
        int p = hi * 8 + v;
        int k = ks * 32 + 2 * p;
        a.u[v] = *(const uint32_t*)(hbuf + m * HDIM + k);
    }
    return a.h;
}

template <bool G>
__device__ __forceinline__ void sdf_forward(const WaveCtx& cx,
                                            float px, float py, float pz,
                                            float& s_out, float& t_out, float& r_out) {
    int lane = cx.lane;
    if (lane < 16) {
        cx.pts[lane * 4 + 0] = px;
        cx.pts[lane * 4 + 1] = py;
        cx.pts[lane * 4 + 2] = pz;
    }
    wait_ds();
    // ---- layer 1 (K=3, VALU): lane owns columns 2*lane, 2*lane+1 ----
    int c0 = lane * 2;
#pragma unroll 1
    for (int m = 0; m < 16; ++m) {
        float x = cx.pts[m * 4 + 0], y = cx.pts[m * 4 + 1], z = cx.pts[m * 4 + 2];
        float a0 = fmaf(x, cx.W1[0 * HDIM + c0],
                   fmaf(y, cx.W1[1 * HDIM + c0],
                   fmaf(z, cx.W1[2 * HDIM + c0], cx.b1[c0])));
        float a1 = fmaf(x, cx.W1[0 * HDIM + c0 + 1],
                   fmaf(y, cx.W1[1 * HDIM + c0 + 1],
                   fmaf(z, cx.W1[2 * HDIM + c0 + 1], cx.b1[c0 + 1])));
        H2 pk;
        pk.h[0] = (_Float16)softplusf(a0);
        pk.h[1] = (_Float16)softplusf(a1);
        *(uint32_t*)(cx.h1 + m * HDIM + c0) = pk.u;
        if (G) {
            H2 sg;
            sg.h[0] = (_Float16)sigmoidf_(a0);
            sg.h[1] = (_Float16)sigmoidf_(a1);
            *(uint32_t*)(cx.sp1 + m * HDIM + c0) = sg.u;
        }
    }
    wait_ds();
    // ---- layer 2 (64x64) via WMMA: 4 N-tiles x 2 K-steps ----
    int n0 = lane & 15, hi = lane >> 4;
#pragma unroll 1
    for (int t = 0; t < 4; ++t) {
        v8f acc = {};
        v16h a0 = load_a_frag(cx.h1, 0, lane);
        acc = __builtin_amdgcn_wmma_f32_16x16x32_f16(
            false, a0, false, cx.W2B[(t * 2 + 0) * 32 + lane].h, (short)0, acc, false, false);
        v16h a1 = load_a_frag(cx.h1, 1, lane);
        acc = __builtin_amdgcn_wmma_f32_16x16x32_f16(
            false, a1, false, cx.W2B[(t * 2 + 1) * 32 + lane].h, (short)0, acc, false, false);
        int n = t * 16 + n0;
#pragma unroll
        for (int v = 0; v < 8; ++v) {
            int m = hi * 8 + v;            // C/D layout: lanes 0-15 -> M=v, 16-31 -> M=8+v
            float val = acc[v] + cx.b2[n];
            cx.h2[m * HDIM + n] = (_Float16)softplusf(val);
            if (G) cx.sp2[m * HDIM + n] = (_Float16)sigmoidf_(val);
        }
    }
    wait_ds();
    // ---- layer 3 (64 -> 1) on WMMA: B column 0 holds W3 ----
    {
        v8f acc3 = {};
        v16h a0 = load_a_frag(cx.h2, 0, lane);
        acc3 = __builtin_amdgcn_wmma_f32_16x16x32_f16(
            false, a0, false, cx.W3B[0 * 32 + lane].h, (short)0, acc3, false, false);
        v16h a1 = load_a_frag(cx.h2, 1, lane);
        acc3 = __builtin_amdgcn_wmma_f32_16x16x32_f16(
            false, a1, false, cx.W3B[1 * 32 + lane].h, (short)0, acc3, false, false);
        if (n0 == 0) {
#pragma unroll
            for (int v = 0; v < 8; ++v) cx.red[hi * 8 + v] = acc3[v];
        }
    }
    wait_ds();
    float dot = cx.red[lane & 15];
    float tt = ftanh(dot + cx.b3);
    float r  = sqrtf(px * px + py * py + pz * pz + 1e-12f);
    s_out = r - 0.1f + 0.01f * tt;
    t_out = tt;
    r_out = r;
}

__device__ __forceinline__ float evalAt(const WaveCtx& cx,
                                        float ox, float oy, float oz,
                                        float dx, float dy, float dz, float t) {
    float s, tt, r;
    sdf_forward<false>(cx, fmaf(t, dx, ox), fmaf(t, dy, oy), fmaf(t, dz, oz), s, tt, r);
    return s;
}

// Forward + analytic gradient; backward 64x64 and final 64->3 projection on WMMA.
__device__ __forceinline__ void sdf_grad(const WaveCtx& cx,
                                         float px, float py, float pz,
                                         float& gx, float& gy, float& gz) {
    float s, tt, r;
    sdf_forward<true>(cx, px, py, pz, s, tt, r);
    int lane = cx.lane;
    int n0 = lane & 15, hi = lane >> 4;
    // V2[m][k] = sigmoid(a2)[m][k] * W3[k]  -> reuse h1 buffer
    int c0 = lane * 2;
    float w3a = cx.W3[c0], w3b = cx.W3[c0 + 1];
#pragma unroll 1
    for (int m = 0; m < 16; ++m) {
        H2 pk;
        pk.h[0] = (_Float16)((float)cx.sp2[m * HDIM + c0]     * w3a);
        pk.h[1] = (_Float16)((float)cx.sp2[m * HDIM + c0 + 1] * w3b);
        *(uint32_t*)(cx.h1 + m * HDIM + c0) = pk.u;
    }
    wait_ds();
    // G1 = V2 @ W2^T ; fold sigma'(a1) in epilogue -> h2
#pragma unroll 1
    for (int t = 0; t < 4; ++t) {
        v8f acc = {};
        v16h a0 = load_a_frag(cx.h1, 0, lane);
        acc = __builtin_amdgcn_wmma_f32_16x16x32_f16(
            false, a0, false, cx.W2TB[(t * 2 + 0) * 32 + lane].h, (short)0, acc, false, false);
        v16h a1 = load_a_frag(cx.h1, 1, lane);
        acc = __builtin_amdgcn_wmma_f32_16x16x32_f16(
            false, a1, false, cx.W2TB[(t * 2 + 1) * 32 + lane].h, (short)0, acc, false, false);
        int n = t * 16 + n0;
#pragma unroll
        for (int v = 0; v < 8; ++v) {
            int m = hi * 8 + v;
            cx.h2[m * HDIM + n] = (_Float16)(acc[v] * (float)cx.sp1[m * HDIM + n]);
        }
    }
    wait_ds();
    // grad_mlp = V1 @ W1^T on WMMA (cols 0..2 of B) -> scatter via pts buffer
    {
        v8f accg = {};
        v16h a0 = load_a_frag(cx.h2, 0, lane);
        accg = __builtin_amdgcn_wmma_f32_16x16x32_f16(
            false, a0, false, cx.W1TB[0 * 32 + lane].h, (short)0, accg, false, false);
        v16h a1 = load_a_frag(cx.h2, 1, lane);
        accg = __builtin_amdgcn_wmma_f32_16x16x32_f16(
            false, a1, false, cx.W1TB[1 * 32 + lane].h, (short)0, accg, false, false);
        if (n0 < 3) {
#pragma unroll
            for (int v = 0; v < 8; ++v) cx.pts[(hi * 8 + v) * 4 + n0] = accg[v];
        }
    }
    wait_ds();
    int m = lane & 15;
    float ax = cx.pts[m * 4 + 0], ay = cx.pts[m * 4 + 1], az = cx.pts[m * 4 + 2];
    float scale = 0.01f * (1.f - tt * tt);
    float inv_r = __builtin_amdgcn_rcpf(r);
    gx = fmaf(scale, ax, px * inv_r);
    gy = fmaf(scale, ay, py * inv_r);
    gz = fmaf(scale, az, pz * inv_r);
}

__device__ __forceinline__ void aabb(float ox, float oy, float oz,
                                     float dx, float dy, float dz,
                                     float& tn, float& tf, bool& hit) {
    float lo, hi2, tmin, tmax;
    {
        float dd = (fabsf(dx) < 1e-9f) ? 1e-9f : dx; float inv = 1.f / dd;
        float t1 = (-0.15f - ox) * inv, t2 = (0.15f - ox) * inv;
        tmin = fminf(t1, t2); tmax = fmaxf(t1, t2);
    }
    {
        float dd = (fabsf(dy) < 1e-9f) ? 1e-9f : dy; float inv = 1.f / dd;
        float t1 = (-0.15f - oy) * inv, t2 = (0.15f - oy) * inv;
        lo = fminf(t1, t2); hi2 = fmaxf(t1, t2);
        tmin = fmaxf(tmin, lo); tmax = fminf(tmax, hi2);
    }
    {
        float dd = (fabsf(dz) < 1e-9f) ? 1e-9f : dz; float inv = 1.f / dd;
        float t1 = (-0.15f - oz) * inv, t2 = (0.15f - oz) * inv;
        lo = fminf(t1, t2); hi2 = fmaxf(t1, t2);
        tmin = fmaxf(tmin, lo); tmax = fminf(tmax, hi2);
    }
    hit = (tmax >= tmin) && (tmax > 0.f);
    tn = hit ? fmaxf(tmin, 0.f) : BIGT;
    tf = hit ? tmax : BIGT;
}

template <bool FAR>
__device__ __forceinline__ void intersect_sdf(const WaveCtx& cx,
                                              float ox, float oy, float oz,
                                              float dx, float dy, float dz,
                                              bool mask, float t_near, float t_far,
                                              bool& valid, float& depth,
                                              float& nx, float& ny, float& nz) {
    float acc_s = mask ? t_near : 0.f;
    float acc_e = mask ? t_far  : 0.f;
    float vs = evalAt(cx, ox, oy, oz, dx, dy, dz, acc_s);
    float ve = evalAt(cx, ox, oy, oz, dx, dy, dz, acc_e);
    bool unf_s = mask && (fabsf(vs) > SDF_THR);
    bool unf_e = mask && (fabsf(ve) > SDF_THR);
#pragma unroll 1
    for (int it = 0; it < 15; ++it) {
        float cs = evalAt(cx, ox, oy, oz, dx, dy, dz, acc_s);
        float ce = evalAt(cx, ox, oy, oz, dx, dy, dz, acc_e);
        acc_s += unf_s ? cs : 0.f;
        acc_e -= unf_e ? ce : 0.f;
        bool nc = acc_s < acc_e;
        unf_s = unf_s && (fabsf(cs) > SDF_THR) && nc;
        unf_e = unf_e && (fabsf(ce) > SDF_THR) && nc;
    }
    bool net_mask = acc_s < acc_e;
    bool sampler_mask = FAR ? unf_e : unf_s;

    float t_lo = acc_s, t_hi = acc_e;
    float prev_t = 0.f, prev_v = 0.f;
    int firstNeg = -1, lastNeg = -1;
    bool pendB = false;
    float t_a = t_lo, f_a = 0.f, t_b = t_lo, f_b = 0.f;
#pragma unroll 1
    for (int si = 0; si < 64; ++si) {
        float ts = fmaf((float)si * (1.f / 63.f), t_hi - t_lo, t_lo);
        float v = evalAt(cx, ox, oy, oz, dx, dy, dz, ts);
        if (!FAR) {
            if (firstNeg < 0 && v < 0.f) {
                firstNeg = si;
                t_a = (si > 0) ? prev_t : ts;
                f_a = (si > 0) ? prev_v : v;
                t_b = ts; f_b = v;
            }
        } else {
            if (v < 0.f) { lastNeg = si; t_a = ts; f_a = v; t_b = ts; f_b = v; pendB = true; }
            else if (pendB) { t_b = ts; f_b = v; pendB = false; }
        }
        prev_t = ts; prev_v = v;
    }
    bool has = FAR ? ((lastNeg >= 0) && (lastNeg < 63) && sampler_mask)
                   : ((firstNeg > 0) && sampler_mask);

#pragma unroll 1
    for (int it = 0; it < 8; ++it) {
        float den = f_b - f_a;
        den = (fabsf(den) < 1e-12f) ? 1e-12f : den;
        float t_mid = t_a - f_a * (t_b - t_a) / den;
        float f_mid = evalAt(cx, ox, oy, oz, dx, dy, dz, t_mid);
        bool same_a = (f_mid > 0.f) == (f_a > 0.f);
        float ta2 = same_a ? t_mid : t_a, fa2 = same_a ? f_mid : f_a;
        float tb2 = same_a ? t_b : t_mid, fb2 = same_a ? f_b : f_mid;
        t_a = ta2; f_a = fa2; t_b = tb2; f_b = fb2;
    }
    float den = f_b - f_a;
    den = (fabsf(den) < 1e-12f) ? 1e-12f : den;
    float t_pred = t_a - f_a * (t_b - t_a) / den;

    depth = sampler_mask ? t_pred : (FAR ? acc_e : acc_s);
    net_mask = sampler_mask ? has : net_mask;
    valid = net_mask && mask;

    float hx = fmaf(depth, dx, ox), hy = fmaf(depth, dy, oy), hz = fmaf(depth, dz, oz);
    float gx, gy, gz;
    sdf_grad(cx, hx, hy, hz, gx, gy, gz);
    float len = sqrtf(gx * gx + gy * gy + gz * gz) + 1e-12f;
    float il = __builtin_amdgcn_rcpf(len);
    nx = valid ? gx * il : 0.f;
    ny = valid ? gy * il : 0.f;
    nz = valid ? gz * il : 0.f;
}

__device__ __forceinline__ void refract3(float lx, float ly, float lz,
                                         float nx, float ny, float nz,
                                         float eta1, float eta2,
                                         float& tx, float& ty, float& tz, float& att) {
    float cosI = -(lx * nx + ly * ny + lz * nz);
    float ipx = fmaf(nx, cosI, lx), ipy = fmaf(ny, cosI, ly), ipz = fmaf(nz, cosI, lz);
    float k = eta1 / eta2;
    float tpx = k * ipx, tpy = k * ipy, tpz = k * ipz;
    float tpn = tpx * tpx + tpy * tpy + tpz * tpz;
    float cl = fminf(fmaxf(tpn, 0.f), 0.999999f);
    float ti = sqrtf(1.f - cl);
    float tx0 = fmaf(-ti, nx, tpx), ty0 = fmaf(-ti, ny, tpy), tz0 = fmaf(-ti, nz, tpz);
    float len = sqrtf(tx0 * tx0 + ty0 * ty0 + tz0 * tz0) + 1e-10f;
    float il = __builtin_amdgcn_rcpf(len);
    tx = tx0 * il; ty = ty0 * il; tz = tz0 * il;
    float cosT = -(tx * nx + ty * ny + tz * nz);
    float e_i = (cosT * eta1 - cosI * eta2) / (cosT * eta1 + cosI * eta2 + 1e-10f);
    float e_p = (cosT * eta2 - cosI * eta1) / (cosT * eta2 + cosI * eta1 + 1e-10f);
    att = fminf(fmaxf(0.5f * (e_i * e_i + e_p * e_p), 0.f), 1.f);
}

__global__ __launch_bounds__(128) void trace_sdf_kernel(
    const float* __restrict__ origins, const float* __restrict__ directions,
    const float* __restrict__ W1, const float* __restrict__ b1,
    const float* __restrict__ W2, const float* __restrict__ b2,
    const float* __restrict__ W3, const float* __restrict__ b3,
    float* __restrict__ out, int N) {

    __shared__ FragU sW2B[4 * 2 * 32];
    __shared__ FragU sW2TB[4 * 2 * 32];
    __shared__ FragU sW3B[2 * 32];
    __shared__ FragU sW1TB[2 * 32];
    __shared__ float sW1[3 * HDIM];
    __shared__ float sb1[HDIM], sb2[HDIM], sW3[HDIM];
    __shared__ float     sPts[WAVES_PER_BLOCK][16 * 4];
    __shared__ float     sRed[WAVES_PER_BLOCK][16];
    __shared__ _Float16  sH1[WAVES_PER_BLOCK][16 * HDIM];
    __shared__ _Float16  sH2[WAVES_PER_BLOCK][16 * HDIM];
    __shared__ _Float16  sSp1[WAVES_PER_BLOCK][16 * HDIM];
    __shared__ _Float16  sSp2[WAVES_PER_BLOCK][16 * HDIM];

    int tid = threadIdx.x;
    for (int i = tid; i < 3 * HDIM; i += blockDim.x) sW1[i] = W1[i];
    for (int i = tid; i < HDIM; i += blockDim.x) { sb1[i] = b1[i]; sb2[i] = b2[i]; sW3[i] = W3[i]; }
    for (int f = tid; f < 4 * 2 * 32; f += blockDim.x) {
        int lane = f & 31;
        int ks   = (f >> 5) & 1;
        int t    = f >> 6;
        int n    = t * 16 + (lane & 15);
        int hi   = lane >> 4;
        FragU fb, ft;
#pragma unroll
        for (int v = 0; v < 8; ++v) {
            int k = ks * 32 + hi * 16 + 2 * v;
            H2 p0, p1;
            p0.h[0] = (_Float16)W2[k * HDIM + n];
            p0.h[1] = (_Float16)W2[(k + 1) * HDIM + n];
            p1.h[0] = (_Float16)W2[n * HDIM + k];
            p1.h[1] = (_Float16)W2[n * HDIM + k + 1];
            fb.u[v] = p0.u; ft.u[v] = p1.u;
        }
        sW2B[f] = fb; sW2TB[f] = ft;
    }
    for (int f = tid; f < 2 * 32; f += blockDim.x) {
        int lane = f & 31;
        int ks   = f >> 5;
        int n0   = lane & 15;
        int hi   = lane >> 4;
        FragU f3, f1;
#pragma unroll
        for (int v = 0; v < 8; ++v) {
            int k = ks * 32 + hi * 16 + 2 * v;
            H2 p3, p1;
            p3.h[0] = (n0 == 0) ? (_Float16)W3[k]     : (_Float16)0.f;
            p3.h[1] = (n0 == 0) ? (_Float16)W3[k + 1] : (_Float16)0.f;
            p1.h[0] = (n0 < 3) ? (_Float16)W1[n0 * HDIM + k]     : (_Float16)0.f;
            p1.h[1] = (n0 < 3) ? (_Float16)W1[n0 * HDIM + k + 1] : (_Float16)0.f;
            f3.u[v] = p3.u; f1.u[v] = p1.u;
        }
        sW3B[f] = f3; sW1TB[f] = f1;
    }
    __syncthreads();

    int wv = tid >> 5, lane = tid & 31;
    WaveCtx cx{ sW1, sb1, sb2, sW3, b3[0], sW2B, sW2TB, sW3B, sW1TB,
                sPts[wv], sRed[wv], sH1[wv], sH2[wv], sSp1[wv], sSp2[wv], lane };

    int ray = blockIdx.x * RAYS_PER_BLOCK + wv * 16 + (lane & 15);
    bool wr = (lane < 16) && (ray < N);
    int rr = ray < N ? ray : N - 1;

    float ox = origins[rr * 3 + 0], oy = origins[rr * 3 + 1], oz = origins[rr * 3 + 2];
    float dx = directions[rr * 3 + 0], dy = directions[rr * 3 + 1], dz = directions[rr * 3 + 2];

    float tn1, tf1; bool hit1;
    aabb(ox, oy, oz, dx, dy, dz, tn1, tf1, hit1);
    bool valid1; float depth1, n1x, n1y, n1z;
    intersect_sdf<false>(cx, ox, oy, oz, dx, dy, dz, hit1, tn1, tf1,
                         valid1, depth1, n1x, n1y, n1z);

    float p1x = fmaf(depth1, dx, ox), p1y = fmaf(depth1, dy, oy), p1z = fmaf(depth1, dz, oz);
    bool inside = (p1x >= -0.15f) && (p1x <= 0.15f) &&
                  (p1y >= -0.15f) && (p1y <= 0.15f) &&
                  (p1z >= -0.15f) && (p1z <= 0.15f);

    float lt1x, lt1y, lt1z, att1;
    refract3(dx, dy, dz, n1x, n1y, n1z, 1.0003f, 1.45f, lt1x, lt1y, lt1z, att1);
    float rdn = dx * n1x + dy * n1y + dz * n1z;
    float lr1x = fmaf(-2.f * rdn, n1x, dx);
    float lr1y = fmaf(-2.f * rdn, n1y, dy);
    float lr1z = fmaf(-2.f * rdn, n1z, dz);
    float rl = sqrtf(lr1x * lr1x + lr1y * lr1y + lr1z * lr1z) + 1e-10f;
    float irl = __builtin_amdgcn_rcpf(rl);
    lr1x *= irl; lr1y *= irl; lr1z *= irl;

    float o2x = p1x - lt1x, o2y = p1y - lt1y, o2z = p1z - lt1z;
    float tn2, tf2; bool hit2;
    aabb(o2x, o2y, o2z, lt1x, lt1y, lt1z, tn2, tf2, hit2);
    bool mask2 = hit2 && inside && valid1;
    bool valid2; float depth2, n2x, n2y, n2z;
    intersect_sdf<true>(cx, o2x, o2y, o2z, lt1x, lt1y, lt1z, mask2, tn2, tf2,
                        valid2, depth2, n2x, n2y, n2z);

    float p2x = fmaf(depth2, lt1x, o2x), p2y = fmaf(depth2, lt1y, o2y), p2z = fmaf(depth2, lt1z, o2z);
    float lt2x, lt2y, lt2z, att2d;
    refract3(lt1x, lt1y, lt1z, -n2x, -n2y, -n2z, 1.45f, 1.0003f, lt2x, lt2y, lt2z, att2d);

    bool fin = valid1 && valid2;
    if (wr) {
        float* out_o  = out;
        float* out_d  = out + 3 * (size_t)N;
        float* refl_o = out + 6 * (size_t)N;
        float* refl_d = out + 9 * (size_t)N;
        float* attp   = out + 12 * (size_t)N;
        float* finp   = out + 13 * (size_t)N;
        float* fpts   = out + 14 * (size_t)N;
        float* spts   = out + 17 * (size_t)N;
        out_o[ray * 3 + 0] = fin ? fmaf(0.1f, lt2x, p2x) : ox;
        out_o[ray * 3 + 1] = fin ? fmaf(0.1f, lt2y, p2y) : oy;
        out_o[ray * 3 + 2] = fin ? fmaf(0.1f, lt2z, p2z) : oz;
        out_d[ray * 3 + 0] = fin ? lt2x : dx;
        out_d[ray * 3 + 1] = fin ? lt2y : dy;
        out_d[ray * 3 + 2] = fin ? lt2z : dz;
        refl_o[ray * 3 + 0] = fin ? fmaf(0.1f, lr1x, p1x) : ox;
        refl_o[ray * 3 + 1] = fin ? fmaf(0.1f, lr1y, p1y) : oy;
        refl_o[ray * 3 + 2] = fin ? fmaf(0.1f, lr1z, p1z) : oz;
        refl_d[ray * 3 + 0] = fin ? lr1x : dx;
        refl_d[ray * 3 + 1] = fin ? lr1y : dy;
        refl_d[ray * 3 + 2] = fin ? lr1z : dz;
        attp[ray] = fin ? att1 : 0.f;
        finp[ray] = fin ? 1.f : 0.f;
        fpts[ray * 3 + 0] = p1x; fpts[ray * 3 + 1] = p1y; fpts[ray * 3 + 2] = p1z;
        spts[ray * 3 + 0] = p2x; spts[ray * 3 + 1] = p2y; spts[ray * 3 + 2] = p2z;
    }
}

extern "C" void kernel_launch(void* const* d_in, const int* in_sizes, int n_in,
                              void* d_out, int out_size, void* d_ws, size_t ws_size,
                              hipStream_t stream) {
    const float* origins    = (const float*)d_in[0];
    const float* directions = (const float*)d_in[1];
    const float* W1 = (const float*)d_in[2];
    const float* b1 = (const float*)d_in[3];
    const float* W2 = (const float*)d_in[4];
    const float* b2 = (const float*)d_in[5];
    const float* W3 = (const float*)d_in[6];
    const float* b3 = (const float*)d_in[7];
    int N = in_sizes[0] / 3;
    int blocks = (N + RAYS_PER_BLOCK - 1) / RAYS_PER_BLOCK;
    trace_sdf_kernel<<<blocks, 128, 0, stream>>>(origins, directions,
                                                 W1, b1, W2, b2, W3, b3,
                                                 (float*)d_out, N);
}
